// Encoder_68917045231795
// MI455X (gfx1250) — matrix operand order repaired and checked by reading, not compiled, not run on hardware
//
#include <hip/hip_runtime.h>
#include <hip/hip_bf16.h>
#include <math.h>

typedef __attribute__((ext_vector_type(16))) _Float16 v16h;
typedef __attribute__((ext_vector_type(8)))  float    v8f;

#define ND_ 16
#define DD_ 256
#define HH_ 8
#define KK_ 64
#define NB_ 4

__device__ __forceinline__ float gelu_f(float x) {
    const float c = 0.7978845608028654f;
    float x3 = x * x * x;
    return 0.5f * x * (1.0f + tanhf(c * (x + 0.044715f * x3)));
}

__device__ __forceinline__ void atomicMaxF(float* addr, float value) {
    if (value >= 0.0f) atomicMax((int*)addr, __float_as_int(value));
    else               atomicMin((unsigned int*)addr, __float_as_uint(value));
}

// ---------------------------------------------------------------------------
// Pack fp32 weight [Korig, Nn] into fp16 WMMA B-fragment layout, K padded to
// Kpad (multiple of 32). Element (tile, lane, j): k = kt*32 + (lane>>4)*8 + j
// for j < 8, kt*32 + 16 + (lane>>4)*8 + (j-8) for j >= 8,
// col = ct*16 + (lane&15).  (matches 16-bit B 32x16 per-lane layout)
// ---------------------------------------------------------------------------
__global__ __launch_bounds__(256) void pack_b_kernel(
    const float* __restrict__ W, _Float16* __restrict__ out,
    int Korig, int Kpad, int Nn)
{
    int total = Kpad * Nn;
    int ntc = Nn >> 4;
    for (int idx = blockIdx.x * blockDim.x + threadIdx.x; idx < total;
         idx += gridDim.x * blockDim.x) {
        int tile = idx >> 9, within = idx & 511;
        int lane = within >> 4, j = within & 15;
        int kt = tile / ntc, ct = tile - kt * ntc;
        int half = lane >> 4;
        int col = ct * 16 + (lane & 15);
        int k = kt * 32 + ((j < 8) ? (half * 8 + j) : (16 + half * 8 + (j - 8)));
        out[idx] = (k < Korig) ? (_Float16)W[(size_t)k * Nn + col] : (_Float16)0.0f;
    }
}

// ---------------------------------------------------------------------------
// WMMA GEMM: C[M,Nn] = act(A[M,K](f32) x Bpacked[K,Nn](f16) + bias)
// K multiple of 32, Nn multiple of 64. Block = 256 threads = 8 waves.
// Wave w: rows [by*256 + w*32, +32) as TWO A fragments (row clamp, no exec
// masking in the hot loop), cols [bx*64, +64) as 4 B fragments -> 8 WMMAs
// per K-step, each B fragment reused across both A fragments.
// ---------------------------------------------------------------------------
__global__ __launch_bounds__(256) void gemm_wmma_kernel(
    const float* __restrict__ A, const _Float16* __restrict__ Bp,
    const float* __restrict__ bias, float* __restrict__ C,
    int M, int Nn, int K, int act)
{
    const int lane = threadIdx.x & 31;
    const int wid  = threadIdx.x >> 5;
    const int half = lane >> 4;
    const int lr   = lane & 15;
    const int rowBase = blockIdx.y * 256 + wid * 32;
    const int colBase = blockIdx.x * 64;
    const int ntc = Nn >> 4;

    // clamped A rows: out-of-range rows read row M-1; their results are
    // never stored, so no zero-fill / exec masking is needed in the loop.
    int ar0 = rowBase + lr;       if (ar0 > M - 1) ar0 = M - 1;
    int ar1 = rowBase + 16 + lr;  if (ar1 > M - 1) ar1 = M - 1;
    const float* ap0 = A + (size_t)ar0 * K + half * 8;
    const float* ap1 = A + (size_t)ar1 * K + half * 8;

    v8f acc0, acc1, acc2, acc3, acc4, acc5, acc6, acc7;
#pragma unroll
    for (int j = 0; j < 8; ++j) {
        acc0[j]=0.f; acc1[j]=0.f; acc2[j]=0.f; acc3[j]=0.f;
        acc4[j]=0.f; acc5[j]=0.f; acc6[j]=0.f; acc7[j]=0.f;
    }

    for (int k0 = 0; k0 < K; k0 += 32) {
        // ---- A fragments (16x32 f16), converted from f32 on the fly ----
        v16h a0, a1;
        {
            float4 f0 = *(const float4*)(ap0 + k0);
            float4 f1 = *(const float4*)(ap0 + k0 + 4);
            float4 f2 = *(const float4*)(ap0 + k0 + 16);
            float4 f3 = *(const float4*)(ap0 + k0 + 20);
            a0[0]=(_Float16)f0.x; a0[1]=(_Float16)f0.y; a0[2]=(_Float16)f0.z; a0[3]=(_Float16)f0.w;
            a0[4]=(_Float16)f1.x; a0[5]=(_Float16)f1.y; a0[6]=(_Float16)f1.z; a0[7]=(_Float16)f1.w;
            a0[8]=(_Float16)f2.x; a0[9]=(_Float16)f2.y; a0[10]=(_Float16)f2.z; a0[11]=(_Float16)f2.w;
            a0[12]=(_Float16)f3.x; a0[13]=(_Float16)f3.y; a0[14]=(_Float16)f3.z; a0[15]=(_Float16)f3.w;
        }
        {
            float4 f0 = *(const float4*)(ap1 + k0);
            float4 f1 = *(const float4*)(ap1 + k0 + 4);
            float4 f2 = *(const float4*)(ap1 + k0 + 16);
            float4 f3 = *(const float4*)(ap1 + k0 + 20);
            a1[0]=(_Float16)f0.x; a1[1]=(_Float16)f0.y; a1[2]=(_Float16)f0.z; a1[3]=(_Float16)f0.w;
            a1[4]=(_Float16)f1.x; a1[5]=(_Float16)f1.y; a1[6]=(_Float16)f1.z; a1[7]=(_Float16)f1.w;
            a1[8]=(_Float16)f2.x; a1[9]=(_Float16)f2.y; a1[10]=(_Float16)f2.z; a1[11]=(_Float16)f2.w;
            a1[12]=(_Float16)f3.x; a1[13]=(_Float16)f3.y; a1[14]=(_Float16)f3.z; a1[15]=(_Float16)f3.w;
        }
        // ---- B fragments: contiguous 32B loads from packed layout ----
        const _Float16* bb =
            Bp + ((size_t)(k0 >> 5) * ntc + (colBase >> 4)) * 512 + lane * 16;
        v16h b0 = *(const v16h*)(bb);
        v16h b1 = *(const v16h*)(bb + 512);
        v16h b2 = *(const v16h*)(bb + 1024);
        v16h b3 = *(const v16h*)(bb + 1536);
        if (k0 + 32 < K) {
            __builtin_prefetch(bb + (size_t)ntc * 512, 0, 1);  // next K-tile
        }
        acc0 = __builtin_amdgcn_wmma_f32_16x16x32_f16(false, a0, false, b0, (short)0, acc0, false, false);
        acc1 = __builtin_amdgcn_wmma_f32_16x16x32_f16(false, a0, false, b1, (short)0, acc1, false, false);
        acc2 = __builtin_amdgcn_wmma_f32_16x16x32_f16(false, a0, false, b2, (short)0, acc2, false, false);
        acc3 = __builtin_amdgcn_wmma_f32_16x16x32_f16(false, a0, false, b3, (short)0, acc3, false, false);
        acc4 = __builtin_amdgcn_wmma_f32_16x16x32_f16(false, a1, false, b0, (short)0, acc4, false, false);
        acc5 = __builtin_amdgcn_wmma_f32_16x16x32_f16(false, a1, false, b1, (short)0, acc5, false, false);
        acc6 = __builtin_amdgcn_wmma_f32_16x16x32_f16(false, a1, false, b2, (short)0, acc6, false, false);
        acc7 = __builtin_amdgcn_wmma_f32_16x16x32_f16(false, a1, false, b3, (short)0, acc7, false, false);
    }

#define STORE_TILE(ACC, T, RG)                                               \
    do {                                                                     \
        int col = colBase + (T)*16 + lr;                                     \
        float bv = bias ? bias[col] : 0.0f;                                  \
        _Pragma("unroll")                                                    \
        for (int rr = 0; rr < 8; ++rr) {                                     \
            int rw = rowBase + (RG)*16 + rr + half * 8;                      \
            if (rw < M) {                                                    \
                float v = ACC[rr] + bv;                                      \
                if (act == 1) v = gelu_f(v);                                 \
                C[(size_t)rw * Nn + col] = v;                                \
            }                                                                \
        }                                                                    \
    } while (0)

    STORE_TILE(acc0, 0, 0);
    STORE_TILE(acc1, 1, 0);
    STORE_TILE(acc2, 2, 0);
    STORE_TILE(acc3, 3, 0);
    STORE_TILE(acc4, 0, 1);
    STORE_TILE(acc5, 1, 1);
    STORE_TILE(acc6, 2, 1);
    STORE_TILE(acc7, 3, 1);
#undef STORE_TILE
}

// ---------------------------------------------------------------------------
// Build padded node-embed input [N,32] = [node_feature(16) | position(3) | 0]
// ---------------------------------------------------------------------------
__global__ __launch_bounds__(256) void build_hx_kernel(
    const float* __restrict__ nf, const float* __restrict__ pos,
    float* __restrict__ hx, int N)
{
    long total = (long)N * 32;
    for (long idx = (long)blockIdx.x * blockDim.x + threadIdx.x; idx < total;
         idx += (long)gridDim.x * blockDim.x) {
        int row = (int)(idx >> 5), c = (int)(idx & 31);
        float v = 0.0f;
        if (c < ND_)            v = nf[(size_t)row * ND_ + c];
        else if (c < ND_ + 3)   v = pos[(size_t)row * 3 + (c - ND_)];
        hx[idx] = v;
    }
}

__global__ __launch_bounds__(256) void fill_kernel(float* p, float v, long n) {
    for (long i = (long)blockIdx.x * blockDim.x + threadIdx.x; i < n;
         i += (long)gridDim.x * blockDim.x) p[i] = v;
}

// ---------------------------------------------------------------------------
// Fused edge embedding: d, gaussian basis, 2-layer MLP -> eh[E,64].
// Block processes 8 edges per sweep; ee_W1 resident in LDS.
// ---------------------------------------------------------------------------
#define EBT 8
__global__ __launch_bounds__(256) void edge_embed_kernel(
    const float* __restrict__ pos, const float* __restrict__ ef,
    const int* __restrict__ src, const int* __restrict__ dst,
    const float* __restrict__ gmeans, const float* __restrict__ gstds,
    const float* __restrict__ gmul, const float* __restrict__ gbias,
    const float* __restrict__ W1, const float* __restrict__ b1,
    const float* __restrict__ W2, const float* __restrict__ b2,
    float* __restrict__ eh, int E)
{
    __shared__ float sW1[69 * 128];
    __shared__ float sB1[128];
    __shared__ float sIn[EBT][69];
    __shared__ float sHid[EBT][128];
    const int tid = threadIdx.x;
    for (int i = tid; i < 69 * 128; i += 256) sW1[i] = W1[i];
    if (tid < 128) sB1[tid] = b1[tid];
    __syncthreads();
    const float mul = gmul[0], bia = gbias[0];

    for (int e0 = blockIdx.x * EBT; e0 < E; e0 += gridDim.x * EBT) {
        // stage 1: squared distance per edge
        if (tid < EBT) {
            int e = e0 + tid;
            if (e < E) {
                int s = src[e], d = dst[e];
                float dx = pos[s*3+0] - pos[d*3+0];
                float dy = pos[s*3+1] - pos[d*3+1];
                float dz = pos[s*3+2] - pos[d*3+2];
                sIn[tid][68] = dx*dx + dy*dy + dz*dz;
            }
        }
        __syncthreads();
        // stage 2: edge features + gaussian basis
        for (int t = tid; t < EBT * 68; t += 256) {
            int le = t / 68, c = t - le * 68;
            int e = e0 + le;
            if (e < E) {
                float v;
                if (c < 4) v = ef[(size_t)e * 4 + c];
                else {
                    int g = c - 4;
                    float xd = sIn[le][68] * mul + bia;
                    float z = (xd - gmeans[g]) / gstds[g];
                    v = expf(-0.5f * z * z);
                }
                sIn[le][c] = v;
            }
        }
        __syncthreads();
        // stage 3: hidden = gelu(in @ W1 + b1)
        for (int t = tid; t < EBT * 128; t += 256) {
            int le = t >> 7, hid = t & 127;
            float acc = sB1[hid];
            for (int c = 0; c < 69; ++c) acc += sIn[le][c] * sW1[c * 128 + hid];
            sHid[le][hid] = gelu_f(acc);
        }
        __syncthreads();
        // stage 4: out = hidden @ W2 + b2   (W2 streamed from L2)
        for (int t = tid; t < EBT * 64; t += 256) {
            int le = t >> 6, ob = t & 63;
            int e = e0 + le;
            if (e < E) {
                float acc = b2[ob];
                for (int c = 0; c < 128; ++c) acc += sHid[le][c] * W2[c * 64 + ob];
                eh[(size_t)e * 64 + ob] = acc;
            }
        }
        __syncthreads();
    }
}

// ---------------------------------------------------------------------------
// ebias[E, NB*H] = eh @ Web[b] + beb[b] for all NB blocks at once
// ---------------------------------------------------------------------------
__global__ __launch_bounds__(256) void edge_bias_kernel(
    const float* __restrict__ eh, const float* __restrict__ Web,
    const float* __restrict__ beb, float* __restrict__ ebias, int E)
{
    __shared__ float sW[NB_ * 64 * 8];
    __shared__ float sB[NB_ * 8];
    for (int i = threadIdx.x; i < NB_ * 64 * 8; i += 256) sW[i] = Web[i];
    if (threadIdx.x < NB_ * 8) sB[threadIdx.x] = beb[threadIdx.x];
    __syncthreads();
    for (int e = blockIdx.x * blockDim.x + threadIdx.x; e < E;
         e += gridDim.x * blockDim.x) {
        float ev[64];
        const float4* ep = (const float4*)(eh + (size_t)e * 64);
#pragma unroll
        for (int j = 0; j < 16; ++j) {
            float4 t = ep[j];
            ev[4*j]=t.x; ev[4*j+1]=t.y; ev[4*j+2]=t.z; ev[4*j+3]=t.w;
        }
        for (int ob = 0; ob < NB_ * 8; ++ob) {
            float acc = sB[ob];
            int base = (ob >> 3) * 512 + (ob & 7);
            for (int c = 0; c < 64; ++c) acc += ev[c] * sW[base + c * 8];
            ebias[(size_t)e * 32 + ob] = acc;
        }
    }
}

// ---------------------------------------------------------------------------
// Attention: per-(edge, head) score + running segment-max (atomic)
// ---------------------------------------------------------------------------
__global__ __launch_bounds__(256) void attn_score_kernel(
    const float* __restrict__ q, const float* __restrict__ k,
    const int* __restrict__ src, const int* __restrict__ dst,
    const float* __restrict__ ebias, int blk,
    float* __restrict__ score, float* __restrict__ mbuf, int E)
{
    long idx = (long)blockIdx.x * blockDim.x + threadIdx.x;
    if (idx >= (long)E * 8) return;
    int e = (int)(idx >> 3), hh = (int)(idx & 7);
    int s = src[e], d = dst[e];
    const float4* qv = (const float4*)(q + (size_t)d * DD_ + hh * 32);
    const float4* kv = (const float4*)(k + (size_t)s * DD_ + hh * 32);
    float acc = 0.f;
#pragma unroll
    for (int j = 0; j < 8; ++j) {
        float4 a = qv[j], b = kv[j];
        acc += a.x * b.x + a.y * b.y + a.z * b.z + a.w * b.w;
    }
    float sc = acc * 0.17677669529663687f + ebias[(size_t)e * 32 + blk * 8 + hh];
    score[idx] = sc;
    atomicMaxF(&mbuf[(size_t)d * 8 + hh], sc);
}

__global__ __launch_bounds__(256) void softmax_ex_kernel(
    float* __restrict__ score, const float* __restrict__ mbuf,
    float* __restrict__ den, const int* __restrict__ dst, int E)
{
    long idx = (long)blockIdx.x * blockDim.x + threadIdx.x;
    if (idx >= (long)E * 8) return;
    int e = (int)(idx >> 3), hh = (int)(idx & 7);
    int d = dst[e];
    float ex = expf(score[idx] - mbuf[(size_t)d * 8 + hh]);
    score[idx] = ex;
    atomicAdd(&den[(size_t)d * 8 + hh], ex);
}

__global__ __launch_bounds__(256) void scatter_msg_kernel(
    const float* __restrict__ ex, const float* __restrict__ den,
    const float* __restrict__ v, const int* __restrict__ src,
    const int* __restrict__ dst, float* __restrict__ msg, int E)
{
    long idx = (long)blockIdx.x * blockDim.x + threadIdx.x;
    if (idx >= (long)E * DD_) return;
    int e = (int)(idx >> 8), c = (int)(idx & 255);
    int hh = c >> 5;
    int s = src[e], d = dst[e];
    float alpha = ex[(size_t)e * 8 + hh] / (den[(size_t)d * 8 + hh] + 1e-9f);
    atomicAdd(&msg[(size_t)d * DD_ + c], alpha * v[(size_t)s * DD_ + c]);
}

// ---------------------------------------------------------------------------
// h = LayerNorm(h + o) * g + b, one wave32 per row (D=256, 8 elems/lane)
// ---------------------------------------------------------------------------
__global__ __launch_bounds__(256) void ln_add_kernel(
    float* __restrict__ h, const float* __restrict__ o,
    const float* __restrict__ g, const float* __restrict__ b, int Nrows)
{
    int lane = threadIdx.x & 31, wid = threadIdx.x >> 5;
    int row = blockIdx.x * 8 + wid;
    if (row >= Nrows) return;
    float v[8];
    float s = 0.f;
#pragma unroll
    for (int j = 0; j < 8; ++j) {
        int c = j * 32 + lane;
        v[j] = h[(size_t)row * DD_ + c] + o[(size_t)row * DD_ + c];
        s += v[j];
    }
#pragma unroll
    for (int off = 16; off >= 1; off >>= 1) s += __shfl_xor(s, off, 32);
    float mean = s * (1.0f / DD_);
    float sq = 0.f;
#pragma unroll
    for (int j = 0; j < 8; ++j) { float t = v[j] - mean; sq += t * t; }
#pragma unroll
    for (int off = 16; off >= 1; off >>= 1) sq += __shfl_xor(sq, off, 32);
    float rstd = rsqrtf(sq * (1.0f / DD_) + 1e-5f);
#pragma unroll
    for (int j = 0; j < 8; ++j) {
        int c = j * 32 + lane;
        h[(size_t)row * DD_ + c] = (v[j] - mean) * rstd * g[c] + b[c];
    }
}

// ---------------------------------------------------------------------------
extern "C" void kernel_launch(void* const* d_in, const int* in_sizes, int n_in,
                              void* d_out, int out_size, void* d_ws, size_t ws_size,
                              hipStream_t stream)
{
    // inputs in setup_inputs() dict order
    const float* node_feature = (const float*)d_in[0];
    const float* position     = (const float*)d_in[1];
    const float* edge_feature = (const float*)d_in[2];
    const float* ne_W1 = (const float*)d_in[3];
    const float* ne_b1 = (const float*)d_in[4];
    const float* ne_W2 = (const float*)d_in[5];
    const float* ne_b2 = (const float*)d_in[6];
    const float* gbf_means = (const float*)d_in[7];
    const float* gbf_stds  = (const float*)d_in[8];
    const float* gbf_mul   = (const float*)d_in[9];
    const float* gbf_bias  = (const float*)d_in[10];
    const float* ee_W1 = (const float*)d_in[11];
    const float* ee_b1 = (const float*)d_in[12];
    const float* ee_W2 = (const float*)d_in[13];
    const float* ee_b2 = (const float*)d_in[14];
    const float* Wq = (const float*)d_in[15];
    const float* Wk = (const float*)d_in[16];
    const float* Wv = (const float*)d_in[17];
    const float* Wo = (const float*)d_in[18];
    const float* bq = (const float*)d_in[19];
    const float* bk = (const float*)d_in[20];
    const float* bv = (const float*)d_in[21];
    const float* bo = (const float*)d_in[22];
    const float* Web = (const float*)d_in[23];
    const float* beb = (const float*)d_in[24];
    const float* ln1_g = (const float*)d_in[25];
    const float* ln1_b = (const float*)d_in[26];
    const float* ln2_g = (const float*)d_in[27];
    const float* ln2_b = (const float*)d_in[28];
    const float* Wf1 = (const float*)d_in[29];
    const float* bf1 = (const float*)d_in[30];
    const float* Wf2 = (const float*)d_in[31];
    const float* bf2 = (const float*)d_in[32];
    const float* mean_W = (const float*)d_in[33];
    const float* mean_b = (const float*)d_in[34];
    const float* lv_W = (const float*)d_in[35];
    const float* lv_b = (const float*)d_in[36];
    const int* edge_index = (const int*)d_in[37];

    const int N = in_sizes[1] / 3;
    const int E = in_sizes[2] / 4;
    const int* src = edge_index;
    const int* dst = edge_index + E;

    // ---- workspace carve ----
    char* ws = (char*)d_ws;
    size_t off = 0;
    auto take = [&](size_t bytes) {
        size_t p = off;
        off += (bytes + 255) & ~(size_t)255;
        return p;
    };
    float* h     = (float*)(ws + take((size_t)N * 256 * 4));
    float* h1    = (float*)(ws + take((size_t)N * 1024 * 4));   // shared hidden
    float* hx    = (float*)(ws + take((size_t)N * 32 * 4));
    float* eh    = (float*)(ws + take((size_t)E * 64 * 4));
    float* ebias = (float*)(ws + take((size_t)E * 32 * 4));
    float* qb    = (float*)(ws + take((size_t)N * 256 * 4));
    float* kb    = (float*)(ws + take((size_t)N * 256 * 4));
    float* vb    = (float*)(ws + take((size_t)N * 256 * 4));
    float* msg   = (float*)(ws + take((size_t)N * 256 * 4));
    float* obuf  = (float*)(ws + take((size_t)N * 256 * 4));
    float* mbuf  = (float*)(ws + take((size_t)N * 8 * 4));
    float* den   = (float*)(ws + take((size_t)N * 8 * 4));
    float* score = (float*)(ws + take((size_t)E * 8 * 4));

    // packed fp16 weights
    _Float16* p_neW1 = (_Float16*)(ws + take((size_t)32 * 1024 * 2));
    _Float16* p_neW2 = (_Float16*)(ws + take((size_t)1024 * 256 * 2));
    _Float16* p_q    = (_Float16*)(ws + take((size_t)NB_ * 256 * 256 * 2));
    _Float16* p_k    = (_Float16*)(ws + take((size_t)NB_ * 256 * 256 * 2));
    _Float16* p_v    = (_Float16*)(ws + take((size_t)NB_ * 256 * 256 * 2));
    _Float16* p_o    = (_Float16*)(ws + take((size_t)NB_ * 256 * 256 * 2));
    _Float16* p_f1   = (_Float16*)(ws + take((size_t)NB_ * 256 * 1024 * 2));
    _Float16* p_f2   = (_Float16*)(ws + take((size_t)NB_ * 1024 * 256 * 2));
    _Float16* p_mean = (_Float16*)(ws + take((size_t)256 * 64 * 2));
    _Float16* p_lv   = (_Float16*)(ws + take((size_t)256 * 64 * 2));
    (void)ws_size; (void)n_in; (void)out_size;

    auto pack = [&](const float* W, _Float16* out, int Korig, int Kpad, int Nn) {
        int total = Kpad * Nn;
        int blocks = (total + 255) / 256;
        if (blocks > 4096) blocks = 4096;
        pack_b_kernel<<<blocks, 256, 0, stream>>>(W, out, Korig, Kpad, Nn);
    };
    auto gemm = [&](const float* A, const _Float16* B, const float* bias, float* C,
                    int M, int Nn, int K, int act) {
        dim3 g(Nn / 64, (M + 255) / 256);
        gemm_wmma_kernel<<<g, 256, 0, stream>>>(A, B, bias, C, M, Nn, K, act);
    };
    auto fill = [&](float* p, float v, long n) {
        int blocks = (int)((n + 255) / 256);
        if (blocks > 8192) blocks = 8192;
        fill_kernel<<<blocks, 256, 0, stream>>>(p, v, n);
    };

    // ---- pack all GEMM weights to fp16 fragment layout ----
    pack(ne_W1, p_neW1, ND_ + 3, 32, 1024);
    pack(ne_W2, p_neW2, 1024, 1024, 256);
    for (int b = 0; b < NB_; ++b) {
        pack(Wq  + (size_t)b * 256 * 256,  p_q  + (size_t)b * 256 * 256,  256, 256, 256);
        pack(Wk  + (size_t)b * 256 * 256,  p_k  + (size_t)b * 256 * 256,  256, 256, 256);
        pack(Wv  + (size_t)b * 256 * 256,  p_v  + (size_t)b * 256 * 256,  256, 256, 256);
        pack(Wo  + (size_t)b * 256 * 256,  p_o  + (size_t)b * 256 * 256,  256, 256, 256);
        pack(Wf1 + (size_t)b * 256 * 1024, p_f1 + (size_t)b * 256 * 1024, 256, 256, 1024);
        pack(Wf2 + (size_t)b * 1024 * 256, p_f2 + (size_t)b * 1024 * 256, 1024, 1024, 256);
    }
    pack(mean_W, p_mean, 256, 256, 64);
    pack(lv_W,   p_lv,   256, 256, 64);

    // ---- node embedding MLP (WMMA) ----
    {
        int blocks = (int)(((long)N * 32 + 255) / 256);
        if (blocks > 8192) blocks = 8192;
        build_hx_kernel<<<blocks, 256, 0, stream>>>(node_feature, position, hx, N);
    }
    gemm(hx, p_neW1, ne_b1, h1, N, 1024, 32, 1);
    gemm(h1, p_neW2, ne_b2, h,  N, 256, 1024, 0);

    // ---- fused edge embedding + per-block edge bias ----
    edge_embed_kernel<<<1024, 256, 0, stream>>>(
        position, edge_feature, src, dst, gbf_means, gbf_stds, gbf_mul, gbf_bias,
        ee_W1, ee_b1, ee_W2, ee_b2, eh, E);
    edge_bias_kernel<<<1024, 256, 0, stream>>>(eh, Web, beb, ebias, E);

    // ---- DMT blocks ----
    const long eh8 = (long)E * 8;
    const long ed  = (long)E * DD_;
    for (int b = 0; b < NB_; ++b) {
        gemm(h, p_q + (size_t)b * 256 * 256, bq + b * 256, qb, N, 256, 256, 0);
        gemm(h, p_k + (size_t)b * 256 * 256, bk + b * 256, kb, N, 256, 256, 0);
        gemm(h, p_v + (size_t)b * 256 * 256, bv + b * 256, vb, N, 256, 256, 0);

        fill(mbuf, -3.402823466e38f, (long)N * 8);
        fill(den, 0.0f, (long)N * 8);
        fill(msg, 0.0f, (long)N * 256);

        attn_score_kernel<<<(int)((eh8 + 255) / 256), 256, 0, stream>>>(
            qb, kb, src, dst, ebias, b, score, mbuf, E);
        softmax_ex_kernel<<<(int)((eh8 + 255) / 256), 256, 0, stream>>>(
            score, mbuf, den, dst, E);
        scatter_msg_kernel<<<(int)((ed + 255) / 256), 256, 0, stream>>>(
            score, den, vb, src, dst, msg, E);

        gemm(msg, p_o + (size_t)b * 256 * 256, bo + b * 256, obuf, N, 256, 256, 0);
        ln_add_kernel<<<(N + 7) / 8, 256, 0, stream>>>(
            h, obuf, ln1_g + b * 256, ln1_b + b * 256, N);

        gemm(h,  p_f1 + (size_t)b * 256 * 1024, bf1 + b * 1024, h1,   N, 1024, 256, 1);
        gemm(h1, p_f2 + (size_t)b * 1024 * 256, bf2 + b * 256,  obuf, N, 256, 1024, 0);
        ln_add_kernel<<<(N + 7) / 8, 256, 0, stream>>>(
            h, obuf, ln2_g + b * 256, ln2_b + b * 256, N);
    }

    // ---- output heads ----
    float* out_mean = (float*)d_out;
    float* out_lv   = out_mean + (size_t)N * 64;
    gemm(h, p_mean, mean_b, out_mean, N, 64, 256, 0);
    gemm(h, p_lv,   lv_b,   out_lv,   N, 64, 256, 0);
}
